// Get_Abstract_Caps_Up_81733227643430
// MI455X (gfx1250) — matrix operand order repaired and checked by reading, not compiled, not run on hardware
//
#include <hip/hip_runtime.h>

typedef __attribute__((ext_vector_type(16))) _Float16 v16h;
typedef __attribute__((ext_vector_type(8)))  _Float16 v8h;
typedef __attribute__((ext_vector_type(8)))  float    v8f;

#define LDA 40  // f16 elements per LDS row: 32 K + 8 pad (row stride 80B, 16B aligned)

// ---------------- CDNA5 async-to-LDS helpers (inline asm, ASYNCcnt-tracked) ----------
__device__ __forceinline__ uint32_t lds_off32(const void* p) {
  // generic 'shared' address: low 32 bits are the LDS byte address
  return (uint32_t)(uintptr_t)p;
}
__device__ __forceinline__ void async_ld_b128x2(uint32_t lds, const void* g) {
  // INST_OFFSET is added to BOTH the LDS and global address (ISA 15.18 async pseudocode)
  asm volatile("global_load_async_to_lds_b128 %0, %1, off\n\t"
               "global_load_async_to_lds_b128 %0, %1, off offset:16"
               :: "v"(lds), "v"(g) : "memory");
}
__device__ __forceinline__ void wait_async0() {
  asm volatile("s_wait_asynccnt 0x0" ::: "memory");
}

// ---------------- prep 1: x f32 -> f16, padded 33x33 with zero guard row/col ----------
// xf layout: [n][mi][ih 0..32][iw 0..32][di 0..15] f16  (granule = 16 halfs = 32B)
__global__ __launch_bounds__(256) void prep_x_f16(const float* __restrict__ x,
                                                  _Float16* __restrict__ xf) {
  const int g = blockIdx.x * 256 + threadIdx.x;      // 16*16*33*33 = 278784 granules
  const int iw = g % 33;
  int t1 = g / 33;
  const int ih = t1 % 33;  t1 /= 33;
  const int mi = t1 & 15;
  const int n  = t1 >> 4;
  union { _Float16 h[16]; v8h v[2]; } u;
  if (ih < 32 && iw < 32) {
    const float4* s4 =
        reinterpret_cast<const float4*>(x + ((((n * 16 + mi) * 32 + ih) * 32 + iw) << 4));
    const float4 f0 = s4[0], f1 = s4[1], f2 = s4[2], f3 = s4[3];
    u.h[0]=(_Float16)f0.x; u.h[1]=(_Float16)f0.y; u.h[2]=(_Float16)f0.z; u.h[3]=(_Float16)f0.w;
    u.h[4]=(_Float16)f1.x; u.h[5]=(_Float16)f1.y; u.h[6]=(_Float16)f1.z; u.h[7]=(_Float16)f1.w;
    u.h[8]=(_Float16)f2.x; u.h[9]=(_Float16)f2.y; u.h[10]=(_Float16)f2.z; u.h[11]=(_Float16)f2.w;
    u.h[12]=(_Float16)f3.x; u.h[13]=(_Float16)f3.y; u.h[14]=(_Float16)f3.z; u.h[15]=(_Float16)f3.w;
  } else {
    v8h z = {};
    u.v[0] = z; u.v[1] = z;
  }
  v8h* d = reinterpret_cast<v8h*>(xf + ((size_t)g << 4));
  d[0] = u.v[0]; d[1] = u.v[1];
}

// ---------------- prep 2: weights f32 -> f16, transposed to [tap][co][ci] -------------
__global__ __launch_bounds__(256) void prep_w_f16(const float* __restrict__ w,
                                                  _Float16* __restrict__ w2) {
  const int i = blockIdx.x * 256 + threadIdx.x;      // 9*128*128 = 147456
  const int ci  = i & 127;
  const int co  = (i >> 7) & 127;
  const int tap = i >> 14;
  w2[i] = (_Float16)w[ci * 1152 + co * 9 + tap];     // w[ci][co][kh][kw]
}

// ---------------- main: implicit-GEMM deconv (WMMA f16) + fused bias + squash --------
// ConvTranspose2d: out[co,oh,ow] += x[ci,ih,iw]*w[ci,co,kh,kw], oh = 2*ih - 1 + kh.
// 4 parity classes (py,px): oh=2y+py, ow=2x+px on a 32x32 sub-grid with taps
//   py==0: (ay=0,kh=1)             px==0: (bx=0,kw=1)
//   py==1: (ay=1,kh=0),(ay=0,kh=2) px==1: (bx=1,kw=0),(bx=0,kw=2)
// ih=y+ay, iw=x+bx read from the zero-padded 33x33 f16 copy (no bounds checks).
__global__ __launch_bounds__(256) void caps_deconv_squash_kernel(
    const _Float16* __restrict__ xf,   // [16][16][33][33][16] f16 (padded)
    const _Float16* __restrict__ w2,   // [9][128][128] f16  (tap, co, ci)
    const float* __restrict__ bias,    // [128]
    float* __restrict__ out)           // [16][8][64][64][16]
{
  __shared__ _Float16 As[2][128 * LDA];  // double-buffered A: [m=128][k=32]
  __shared__ _Float16 Bs[2][128 * LDA];  // double-buffered B^T: [co=128][k=32]

  const int bxk = blockIdx.x;
  const int cls = bxk >> 7;            // 4 parity classes
  const int rem = bxk & 127;
  const int n   = rem >> 3;            // batch 0..15
  const int y0  = (rem & 7) << 2;      // 8 tiles of 4 sub-grid rows
  const int py  = cls >> 1, px = cls & 1;

  int ayA[2], khA[2], thCnt;
  if (py == 0) { thCnt = 1; ayA[0] = 0; khA[0] = 1; ayA[1] = 0; khA[1] = 1; }
  else         { thCnt = 2; ayA[0] = 1; khA[0] = 0; ayA[1] = 0; khA[1] = 2; }
  int bxA[2], kwA[2], twCnt;
  if (px == 0) { twCnt = 1; bxA[0] = 0; kwA[0] = 1; bxA[1] = 0; kwA[1] = 1; }
  else         { twCnt = 2; bxA[0] = 1; kwA[0] = 0; bxA[1] = 0; kwA[1] = 2; }

  // flattened tap list
  int ayT[4], bxT[4], offT[4], nTap = 0;
  for (int ti = 0; ti < thCnt; ++ti)
    for (int tj = 0; tj < twCnt; ++tj) {
      ayT[nTap]  = ayA[ti];
      bxT[nTap]  = bxA[tj];
      offT[nTap] = khA[ti] * 3 + kwA[tj];
      ++nTap;
    }
  const int T = nTap << 2;             // taps * 4 K-chunks of 32

  const int t    = threadIdx.x;
  const int lane = t & 31;
  const int wid  = t >> 5;             // 8 waves: 4 (M) x 2 (N)
  const int m0w  = (wid & 3) << 5;     // 0,32,64,96
  const int n0w  = (wid >> 2) << 6;    // 0,64
  const int l15  = lane & 15;
  const int lhi  = lane >> 4;

  // staging coordinates (256 threads stage 128x32 f16 for A and for B)
  const int mS   = t & 127;            // A row (position) / B row (=co)
  const int kh16 = t >> 7;             // which 16-wide K half
  const int yS   = y0 + (mS >> 5);
  const int xS   = mS & 31;
  const uint32_t ldsAbase = lds_off32(&As[0][mS * LDA + (kh16 << 4)]);
  const uint32_t ldsBbase = lds_off32(&Bs[0][mS * LDA + (kh16 << 4)]);
  const uint32_t bufStep  = (uint32_t)(128 * LDA * 2);  // bytes between buffers

  v8f acc[2][4];
  {
    v8f z = {};
    #pragma unroll
    for (int i = 0; i < 2; ++i)
      #pragma unroll
      for (int j = 0; j < 4; ++j) acc[i][j] = z;
  }

  // stage(it) into buffer b: pure async f16 copies, no VALU conversion
  auto stage = [&](int it, int b) {
    const int tl = it >> 2, c = it & 3;
    const int ih = yS + ayT[tl];
    const int iw = xS + bxT[tl];
    const int mi = (c << 1) + kh16;
    const _Float16* ga = xf + ((size_t)((((n * 16 + mi) * 33 + ih) * 33 + iw)) << 4);
    async_ld_b128x2(ldsAbase + (uint32_t)b * bufStep, ga);
    const _Float16* gb = w2 + (((offT[tl] << 7) + mS) << 7) + (c << 5) + (kh16 << 4);
    async_ld_b128x2(ldsBbase + (uint32_t)b * bufStep, gb);
  };

  stage(0, 0);
  for (int it = 0; it < T; ++it) {
    const int cur = it & 1;
    wait_async0();        // my async writes for stage(it) landed in LDS
    __syncthreads();      // everyone's staging visible; everyone done reading buf[1-cur]
    if (it + 1 < T) stage(it + 1, 1 - cur);  // DMA next chunk while we compute

    // ---- fragment loads (documented 16-bit A/B VGPR layouts) ----
    v16h af[2], bf[4];
    #pragma unroll
    for (int mt = 0; mt < 2; ++mt) {
      const int m    = m0w + (mt << 4) + l15;
      const int koff = lhi << 3;                 // lanes 0-15: K0-7, 16-31: K8-15
      union { v16h v; v8h h[2]; } u;
      u.h[0] = *reinterpret_cast<const v8h*>(&As[cur][m * LDA + koff]);       // VGPR0-3
      u.h[1] = *reinterpret_cast<const v8h*>(&As[cur][m * LDA + 16 + koff]);  // VGPR4-7
      af[mt] = u.v;
    }
    #pragma unroll
    for (int nt = 0; nt < 4; ++nt) {
      const int nn    = n0w + (nt << 4) + l15;
      const int koff2 = lhi << 3;                // lanes 0-15: K0-7, 16-31: K8-15
      union { v16h v; v8h h[2]; } u;
      u.h[0] = *reinterpret_cast<const v8h*>(&Bs[cur][nn * LDA + koff2]);
      u.h[1] = *reinterpret_cast<const v8h*>(&Bs[cur][nn * LDA + 16 + koff2]);
      bf[nt] = u.v;
    }
    #pragma unroll
    for (int mt = 0; mt < 2; ++mt)
      #pragma unroll
      for (int nt = 0; nt < 4; ++nt)
        acc[mt][nt] = __builtin_amdgcn_wmma_f32_16x16x32_f16(
            false, af[mt], false, bf[nt], (short)0, acc[mt][nt], false, false);
  }

  // ---- epilogue: bias + squash over the 16-wide capsule (N = d = lane index) ----
  #pragma unroll
  for (int nt = 0; nt < 4; ++nt) {
    const int coBase = n0w + (nt << 4);
    const float bval = bias[coBase + l15];
    const int mo = coBase >> 4;
    #pragma unroll
    for (int mt = 0; mt < 2; ++mt) {
      #pragma unroll
      for (int r = 0; r < 8; ++r) {
        const float val = acc[mt][nt][r] + bval;
        float sq = val * val;                    // reduce over d = 16 lanes of this half
        sq += __shfl_xor(sq, 1, 32);
        sq += __shfl_xor(sq, 2, 32);
        sq += __shfl_xor(sq, 4, 32);
        sq += __shfl_xor(sq, 8, 32);
        const float factor = sq / ((1.0f + sq) * sqrtf(sq + 1e-7f));
        const float o = val * factor;
        const int mrow = m0w + (mt << 4) + (lhi << 3) + r;  // C layout: M = v + 8*hi
        const int yy = y0 + (mrow >> 5);
        const int xx = mrow & 31;
        const int oh = (yy << 1) + py;
        const int ow = (xx << 1) + px;
        out[(((n * 8 + mo) * 64 + oh) * 64 + ow) * 16 + l15] = o;
      }
    }
  }
}

extern "C" void kernel_launch(void* const* d_in, const int* in_sizes, int n_in,
                              void* d_out, int out_size, void* d_ws, size_t ws_size,
                              hipStream_t stream) {
  const float* x    = (const float*)d_in[0];  // [16,16,32,32,16]
  const float* wgt  = (const float*)d_in[1];  // [128,128,3,3]
  const float* bias = (const float*)d_in[2];  // [128]
  // d_in[3] = b_route: provably unused — softmax weights sum to 1 over the
  // broadcast (size-1) Min axis of pred, so routing leaves out == squash(pred).
  float* out = (float*)d_out;

  // workspace: xf16 padded [16][16][33][33][16] (8,921,088 B) then w2 [9][128][128] f16
  _Float16* xf = (_Float16*)d_ws;
  _Float16* w2 = xf + (size_t)16 * 16 * 33 * 33 * 16;

  prep_x_f16<<<dim3(1089), dim3(256), 0, stream>>>(x, xf);     // 278784 granules
  prep_w_f16<<<dim3(576),  dim3(256), 0, stream>>>(wgt, w2);   // 147456 elems
  // grid = 4 parity classes * 16 batch * 8 y-tiles; each WG: M=128 positions x N=128 ch
  caps_deconv_squash_kernel<<<dim3(512), dim3(256), 0, stream>>>(xf, w2, bias, out);
}
